// HeteroSimplexLayer_28372553957533
// MI455X (gfx1250) — matrix operand order repaired and checked by reading, not compiled, not run on hardware
//
#include <hip/hip_runtime.h>
#include <math.h>

// ---------------- problem constants ----------------
#define NN_   100000
#define DD_   256
#define RR_   4
#define TT_   150000
#define HH_   2
#define DH_   128

typedef __attribute__((ext_vector_type(16))) __bf16 v16bf;
typedef __attribute__((ext_vector_type(8)))  __bf16 v8bf;
typedef __attribute__((ext_vector_type(4)))  __bf16 v4bf;
typedef __attribute__((ext_vector_type(8)))  float  v8f;

// ---------------- small device helpers ----------------
__device__ __forceinline__ float bf2f(__bf16 x) {
    union { unsigned short u; __bf16 h; } c; c.h = x;
    union { unsigned int u; float f; } o; o.u = ((unsigned int)c.u) << 16;
    return o.f;
}
__device__ __forceinline__ __bf16 f2bf(float f) {
    union { float f; unsigned int u; } c; c.f = f;
    unsigned int r = c.u + 0x7FFFu + ((c.u >> 16) & 1u);   // round-to-nearest-even
    union { unsigned short u; __bf16 h; } o; o.u = (unsigned short)(r >> 16);
    return o.h;
}
__device__ __forceinline__ float gelu_f(float x) {
    return 0.5f * x * (1.0f + erff(x * 0.7071067811865475f));
}
__device__ __forceinline__ float wave_sum(float v) {
    #pragma unroll
    for (int m = 16; m > 0; m >>= 1) v += __shfl_xor(v, m, 32);
    return v;
}

// ---------------- f32 -> bf16 convert ----------------
__global__ void cvt_f32_bf16(const float* __restrict__ s, __bf16* __restrict__ d, long n) {
    long i = (long)blockIdx.x * blockDim.x + threadIdx.x;
    long stride = (long)gridDim.x * blockDim.x;
    for (; i < n; i += stride) d[i] = f2bf(s[i]);
}

// ---------------- WMMA bf16 GEMM: C[M,Nout] = A[M,K] * B[Nout,K]^T (+bias)(+gelu) ----------------
// grid = (ceil(M/128), Nout/64), block = 256 (8 waves). Wave computes a 16x64 strip.
// B k-slices (64 x 32 = 4KB) are staged into LDS once per block via CDNA5
// GLOBAL_LOAD_ASYNC_TO_LDS_B128 (ASYNCcnt), double-buffered to overlap with WMMA.
template<int OUT_BF16, int DO_GELU>
__global__ __launch_bounds__(256)
void gemm_nt_bf16(const __bf16* __restrict__ A, const __bf16* __restrict__ B, int ldb,
                  const float* __restrict__ bias,
                  float* __restrict__ Cf, __bf16* __restrict__ Cb,
                  long M, int Nout, int K)
{
    __shared__ __bf16 sB[2][64 * 32];    // [buf][n_local*32 + k_local], row stride 64B

    const int tid  = threadIdx.x;
    const int wave = tid >> 5;
    const int lane = tid & 31;
    const int l15  = lane & 15;
    const int half = lane >> 4;

    const long mbase = (long)blockIdx.x * 128 + (long)wave * 16;
    const int  nbase = blockIdx.y * 64;

    long arow = mbase + l15;
    if (arow >= M) arow = M - 1;              // clamp loads; stores are guarded
    const __bf16* Aptr = A + arow * (long)K;

    // cooperative async B tile fetch: 256 threads x 16B = 4KB slice
    const int brow = tid >> 2;                // 0..63  (n_local)
    const int bk   = (tid & 3) * 8;           // 0,8,16,24 (k_local)
    auto fetchB = [&](int k, int buf) {
        const __bf16* gp = B + (long)(nbase + brow) * ldb + k + bk;
        unsigned loff = (unsigned)(unsigned long long)(const void*)&sB[buf][brow * 32 + bk];
        asm volatile("global_load_async_to_lds_b128 %0, %1, off"
                     :: "v"(loff), "v"((unsigned long long)gp) : "memory");
    };

    v8f acc0 = {}, acc1 = {}, acc2 = {}, acc3 = {};

    fetchB(0, 0);
    asm volatile("s_wait_asynccnt 0" ::: "memory");
    __syncthreads();

    int buf = 0;
    for (int k = 0; k < K; k += 32) {
        if (k + 32 < K) fetchB(k + 32, buf ^ 1);              // prefetch next slice -> other buffer
        if (k + 64 < K) __builtin_prefetch(Aptr + k + 64, 0, 0); // global_prefetch_b8 on A stream

        union { v16bf v; v8bf h[2]; } af;
        af.h[0] = *(const v8bf*)(Aptr + k + half * 16);
        af.h[1] = *(const v8bf*)(Aptr + k + half * 16 + 8);

        const __bf16* sb = &sB[buf][l15 * 32 + half * 16];
        v16bf b0 = *(const v16bf*)(sb);
        v16bf b1 = *(const v16bf*)(sb + 16 * 32);
        v16bf b2 = *(const v16bf*)(sb + 32 * 32);
        v16bf b3 = *(const v16bf*)(sb + 48 * 32);

        acc0 = __builtin_amdgcn_wmma_f32_16x16x32_bf16(false, af.v, false, b0, (short)0, acc0, false, false);
        acc1 = __builtin_amdgcn_wmma_f32_16x16x32_bf16(false, af.v, false, b1, (short)0, acc1, false, false);
        acc2 = __builtin_amdgcn_wmma_f32_16x16x32_bf16(false, af.v, false, b2, (short)0, acc2, false, false);
        acc3 = __builtin_amdgcn_wmma_f32_16x16x32_bf16(false, af.v, false, b3, (short)0, acc3, false, false);

        asm volatile("s_wait_asynccnt 0" ::: "memory");       // own async issue complete
        __syncthreads();                                      // tile published / safe to overwrite
        buf ^= 1;
    }

    v8f accs[4] = { acc0, acc1, acc2, acc3 };
    #pragma unroll
    for (int t = 0; t < 4; ++t) {
        const int col = nbase + t * 16 + l15;
        const float bv = bias ? bias[col] : 0.0f;
        #pragma unroll
        for (int i = 0; i < 8; ++i) {
            long row = mbase + (long)half * 8 + i;
            if (row < M) {
                float v = accs[t][i] + bv;
                if (DO_GELU) v = gelu_f(v);
                if (OUT_BF16) Cb[row * (long)Nout + col] = f2bf(v);
                else          Cf[row * (long)Nout + col] = v;
            }
        }
    }
}

// ---------------- triangle gather + average (exploits linearity of mean before GEMM) ----------------
__global__ void tri_avg_k(const float* __restrict__ x, const int* __restrict__ tri,
                          __bf16* __restrict__ out, int T)
{
    const int t = blockIdx.x;
    const int d = threadIdx.x;
    const long i0 = tri[t], i1 = tri[T + t], i2 = tri[2 * T + t];
    float v = (x[i0 * DD_ + d] + x[i1 * DD_ + d] + x[i2 * DD_ + d]) * (1.0f / 3.0f);
    out[(long)t * DD_ + d] = f2bf(v);
}

// ---------------- scatter-mean accumulation (atomics into L2-resident accumulator) ----------------
__global__ void scatter_k(const float* __restrict__ tri_msg, const int* __restrict__ tri,
                          float* __restrict__ accum, float* __restrict__ cnt, int T)
{
    const int t = blockIdx.x;
    const int d = threadIdx.x;
    const float v = tri_msg[(long)t * DD_ + d];
    #pragma unroll
    for (int vt = 0; vt < 3; ++vt) {
        long node = tri[vt * T + t];
        atomicAdd(accum + node * DD_ + d, v);
    }
    if (d == 0) {
        #pragma unroll
        for (int vt = 0; vt < 3; ++vt) atomicAdd(cnt + tri[vt * T + t], 1.0f);
    }
}

// ---------------- node_msg = accum / max(cnt,1); u = elu(node_msg) ----------------
__global__ void elu_k(const float* __restrict__ accum, const float* __restrict__ cnt,
                      float* __restrict__ u_f, __bf16* __restrict__ u_b)
{
    const long n = blockIdx.x;
    const int d = threadIdx.x;
    const float c = fmaxf(cnt[n], 1.0f);
    const float nm = accum[n * DD_ + d] / c;
    const float u = nm > 0.0f ? nm : (expf(nm) - 1.0f);
    u_f[n * DD_ + d] = u;
    u_b[n * DD_ + d] = f2bf(u);
}

// ---------------- gate combine: a = sigmoid(xg+ug+gb); h = tanh(u)*a + x_res*(1-a) ----------------
__global__ void combine_k(const float* __restrict__ xg, const float* __restrict__ ug,
                          const float* __restrict__ gate_b, const float* __restrict__ u_f,
                          const float* __restrict__ x_res,
                          float* __restrict__ src_f, __bf16* __restrict__ src_b)
{
    const long n = blockIdx.x;
    const int d = threadIdx.x;
    const long i = n * DD_ + d;
    const float z = xg[i] + ug[i] + gate_b[d];
    const float a = 1.0f / (1.0f + expf(-z));
    const float h = tanhf(u_f[i]) * a + x_res[i] * (1.0f - a);
    src_f[i] = h;
    src_b[i] = f2bf(h);
}

// ---------------- tiny 4x4 attention over the relation axis, one wave per (node, head) ----------------
__global__ __launch_bounds__(256)
void attn_k(const __bf16* __restrict__ qkv, __bf16* __restrict__ o)
{
    const long job = (long)blockIdx.x * 8 + (threadIdx.x >> 5);
    if (job >= (long)NN_ * HH_) return;
    const int lane = threadIdx.x & 31;
    const long n = job >> 1;
    const int h = (int)(job & 1);
    const int off = h * DH_ + lane * 4;
    const float scale = 0.08838834764831845f;   // 1/sqrt(128)

    float q[RR_][4], k[RR_][4], v[RR_][4];
    #pragma unroll
    for (int r = 0; r < RR_; ++r) {
        const __bf16* base = qkv + ((long)r * NN_ + n) * (3 * DD_);
        v4bf qv = *(const v4bf*)(base + off);
        v4bf kv = *(const v4bf*)(base + DD_ + off);
        v4bf vv = *(const v4bf*)(base + 2 * DD_ + off);
        #pragma unroll
        for (int e = 0; e < 4; ++e) { q[r][e] = bf2f(qv[e]); k[r][e] = bf2f(kv[e]); v[r][e] = bf2f(vv[e]); }
    }

    float s[RR_][RR_];
    #pragma unroll
    for (int qr = 0; qr < RR_; ++qr)
        #pragma unroll
        for (int kr = 0; kr < RR_; ++kr) {
            float p = q[qr][0]*k[kr][0] + q[qr][1]*k[kr][1] + q[qr][2]*k[kr][2] + q[qr][3]*k[kr][3];
            s[qr][kr] = wave_sum(p) * scale;
        }

    #pragma unroll
    for (int qr = 0; qr < RR_; ++qr) {
        float mx = fmaxf(fmaxf(s[qr][0], s[qr][1]), fmaxf(s[qr][2], s[qr][3]));
        float e0 = expf(s[qr][0]-mx), e1 = expf(s[qr][1]-mx), e2 = expf(s[qr][2]-mx), e3 = expf(s[qr][3]-mx);
        float inv = 1.0f / (e0 + e1 + e2 + e3);
        float a0 = e0*inv, a1 = e1*inv, a2 = e2*inv, a3 = e3*inv;
        __bf16* dst = o + ((long)qr * NN_ + n) * DD_ + off;
        #pragma unroll
        for (int e = 0; e < 4; ++e)
            dst[e] = f2bf(a0*v[0][e] + a1*v[1][e] + a2*v[2][e] + a3*v[3][e]);
    }
}

// ---------------- LN1: x1 = LN(src + o_lin)  (o_lin already has out_b) ----------------
__global__ __launch_bounds__(256)
void ln1_k(const float* __restrict__ src, const float* __restrict__ ol,
           const float* __restrict__ g, const float* __restrict__ b,
           float* __restrict__ x1f, __bf16* __restrict__ x1b, long rows)
{
    const long row = (long)blockIdx.x * 8 + (threadIdx.x >> 5);
    if (row >= rows) return;
    const int lane = threadIdx.x & 31;
    const long base = row * DD_ + lane * 8;

    float y[8]; float s1 = 0.f, s2 = 0.f;
    float4 a0 = *(const float4*)(src + base),     a1 = *(const float4*)(src + base + 4);
    float4 b0 = *(const float4*)(ol + base),      b1 = *(const float4*)(ol + base + 4);
    y[0]=a0.x+b0.x; y[1]=a0.y+b0.y; y[2]=a0.z+b0.z; y[3]=a0.w+b0.w;
    y[4]=a1.x+b1.x; y[5]=a1.y+b1.y; y[6]=a1.z+b1.z; y[7]=a1.w+b1.w;
    #pragma unroll
    for (int i = 0; i < 8; ++i) { s1 += y[i]; s2 += y[i]*y[i]; }
    s1 = wave_sum(s1); s2 = wave_sum(s2);
    const float m = s1 * (1.0f / DD_);
    const float var = s2 * (1.0f / DD_) - m * m;
    const float inv = rsqrtf(var + 1e-5f);
    #pragma unroll
    for (int i = 0; i < 8; ++i) {
        const int d = lane * 8 + i;
        const float v = (y[i] - m) * inv * g[d] + b[d];
        x1f[base + i] = v;
        x1b[base + i] = f2bf(v);
    }
}

// ---------------- LN2 + mean over R -> out ----------------
__global__ __launch_bounds__(256)
void ln2_mean_k(const float* __restrict__ x1f, const float* __restrict__ ffp,
                const float* __restrict__ g, const float* __restrict__ b,
                float* __restrict__ out)
{
    const long n = (long)blockIdx.x * 8 + (threadIdx.x >> 5);
    if (n >= NN_) return;
    const int lane = threadIdx.x & 31;
    float acc[8] = {0,0,0,0,0,0,0,0};

    for (int r = 0; r < RR_; ++r) {
        const long base = ((long)r * NN_ + n) * DD_ + lane * 8;
        float y[8]; float s1 = 0.f, s2 = 0.f;
        float4 a0 = *(const float4*)(x1f + base), a1 = *(const float4*)(x1f + base + 4);
        float4 b0 = *(const float4*)(ffp + base), b1 = *(const float4*)(ffp + base + 4);
        y[0]=a0.x+b0.x; y[1]=a0.y+b0.y; y[2]=a0.z+b0.z; y[3]=a0.w+b0.w;
        y[4]=a1.x+b1.x; y[5]=a1.y+b1.y; y[6]=a1.z+b1.z; y[7]=a1.w+b1.w;
        #pragma unroll
        for (int i = 0; i < 8; ++i) { s1 += y[i]; s2 += y[i]*y[i]; }
        s1 = wave_sum(s1); s2 = wave_sum(s2);
        const float m = s1 * (1.0f / DD_);
        const float inv = rsqrtf(s2 * (1.0f / DD_) - m * m + 1e-5f);
        #pragma unroll
        for (int i = 0; i < 8; ++i) {
            const int d = lane * 8 + i;
            acc[i] += (y[i] - m) * inv * g[d] + b[d];
        }
    }
    const long obase = n * DD_ + lane * 8;
    #pragma unroll
    for (int i = 0; i < 8; ++i) out[obase + i] = acc[i] * (1.0f / RR_);
}

// ================= host side =================
extern "C" void kernel_launch(void* const* d_in, const int* in_sizes, int n_in,
                              void* d_out, int out_size, void* d_ws, size_t ws_size,
                              hipStream_t stream)
{
    (void)in_sizes; (void)n_in; (void)out_size; (void)ws_size;
    const float* x       = (const float*)d_in[0];
    const int*   tris    = (const int*)  d_in[1];
    const float* node_W  = (const float*)d_in[3];
    const float* tri_W   = (const float*)d_in[4];
    const float* res_W   = (const float*)d_in[5];
    const float* gate_W  = (const float*)d_in[6];
    const float* gate_b  = (const float*)d_in[7];
    const float* in_W    = (const float*)d_in[8];
    const float* in_b    = (const float*)d_in[9];
    const float* out_W   = (const float*)d_in[10];
    const float* out_b   = (const float*)d_in[11];
    const float* ln1_g   = (const float*)d_in[12];
    const float* ln1_b   = (const float*)d_in[13];
    const float* lin1_W  = (const float*)d_in[14];
    const float* lin1_b  = (const float*)d_in[15];
    const float* lin2_W  = (const float*)d_in[16];
    const float* lin2_b  = (const float*)d_in[17];
    const float* ln2_g   = (const float*)d_in[18];
    const float* ln2_b   = (const float*)d_in[19];
    float* out = (float*)d_out;

    // ---- workspace bump allocator ----
    char* ws = (char*)d_ws;
    size_t off = 0;
    auto alloc = [&](size_t bytes) -> char* {
        char* p = ws + off;
        off += (bytes + 255) & ~(size_t)255;
        return p;
    };
    const long N = NN_, T = TT_;
    __bf16* x_bf    = (__bf16*)alloc(N * DD_ * 2);
    __bf16* wres    = (__bf16*)alloc((size_t)DD_ * DD_ * 2);
    __bf16* wnode   = (__bf16*)alloc((size_t)RR_ * DD_ * DD_ * 2);
    __bf16* wtri    = (__bf16*)alloc((size_t)RR_ * DD_ * DD_ * 2);
    __bf16* wgate   = (__bf16*)alloc((size_t)DD_ * 2 * DD_ * 2);
    __bf16* win     = (__bf16*)alloc((size_t)3 * DD_ * DD_ * 2);
    __bf16* wout    = (__bf16*)alloc((size_t)DD_ * DD_ * 2);
    __bf16* wlin1   = (__bf16*)alloc((size_t)2 * DD_ * DD_ * 2);
    __bf16* wlin2   = (__bf16*)alloc((size_t)DD_ * 2 * DD_ * 2);
    float*  x_res   = (float*) alloc(N * DD_ * 4);
    float*  xg      = (float*) alloc(N * DD_ * 4);
    __bf16* tri_avg = (__bf16*)alloc(T * DD_ * 2);
    __bf16* tri_ft  = (__bf16*)alloc(T * DD_ * 2);
    float*  tri_msg = (float*) alloc(T * DD_ * 4);
    float*  accum   = (float*) alloc(N * DD_ * 4);
    float*  cnt     = (float*) alloc(N * 4);
    float*  u_f     = (float*) alloc(N * DD_ * 4);
    __bf16* u_b     = (__bf16*)alloc(N * DD_ * 2);
    float*  ug      = (float*) alloc(N * DD_ * 4);
    float*  src_f   = (float*) alloc((size_t)RR_ * N * DD_ * 4);
    __bf16* src_b   = (__bf16*)alloc((size_t)RR_ * N * DD_ * 2);
    __bf16* qkv     = (__bf16*)alloc((size_t)RR_ * N * 3 * DD_ * 2);
    __bf16* o_b     = (__bf16*)alloc((size_t)RR_ * N * DD_ * 2);
    float*  o_lin   = (float*) alloc((size_t)RR_ * N * DD_ * 4);
    float*  x1_f    = (float*) alloc((size_t)RR_ * N * DD_ * 4);
    __bf16* x1_b    = (__bf16*)alloc((size_t)RR_ * N * DD_ * 2);
    __bf16* h1_b    = (__bf16*)alloc((size_t)RR_ * N * 2 * DD_ * 2);
    float*  ffp     = (float*) alloc((size_t)RR_ * N * DD_ * 4);

    auto cvt = [&](const float* s, __bf16* d, long n) {
        int blocks = (int)((n + 255) / 256); if (blocks > 16384) blocks = 16384;
        hipLaunchKernelGGL(cvt_f32_bf16, dim3(blocks), dim3(256), 0, stream, s, d, n);
    };
    // mode: 0 = f32 out, 1 = bf16 out, 2 = bf16 out + gelu
    auto gemm = [&](const __bf16* A, const __bf16* B, int ldb, const float* bias,
                    float* Cf, __bf16* Cb, long M, int Nout, int K, int mode) {
        dim3 grid((unsigned)((M + 127) / 128), (unsigned)(Nout / 64));
        if (mode == 0)      hipLaunchKernelGGL((gemm_nt_bf16<0,0>), grid, dim3(256), 0, stream, A, B, ldb, bias, Cf, Cb, M, Nout, K);
        else if (mode == 1) hipLaunchKernelGGL((gemm_nt_bf16<1,0>), grid, dim3(256), 0, stream, A, B, ldb, bias, Cf, Cb, M, Nout, K);
        else                hipLaunchKernelGGL((gemm_nt_bf16<1,1>), grid, dim3(256), 0, stream, A, B, ldb, bias, Cf, Cb, M, Nout, K);
    };

    // ---- 0. convert inputs / weights to bf16 ----
    cvt(x,      x_bf,  N * DD_);
    cvt(node_W, wnode, (long)RR_ * DD_ * DD_);
    cvt(tri_W,  wtri,  (long)RR_ * DD_ * DD_);
    cvt(res_W,  wres,  (long)DD_ * DD_);
    cvt(gate_W, wgate, (long)DD_ * 2 * DD_);
    cvt(in_W,   win,   (long)3 * DD_ * DD_);
    cvt(out_W,  wout,  (long)DD_ * DD_);
    cvt(lin1_W, wlin1, (long)2 * DD_ * DD_);
    cvt(lin2_W, wlin2, (long)DD_ * 2 * DD_);

    // ---- 1. x_res = x @ res_W.T ; xg = x @ gate_W[:, :D].T (shared across relations) ----
    gemm(x_bf, wres,  DD_,     nullptr, x_res, nullptr, N, DD_, DD_, 0);
    gemm(x_bf, wgate, 2 * DD_, nullptr, xg,    nullptr, N, DD_, DD_, 0);

    // ---- 2. per-relation message passing + gating ----
    for (int r = 0; r < RR_; ++r) {
        const int* tri_r = tris + (size_t)r * 3 * T;
        hipLaunchKernelGGL(tri_avg_k, dim3((unsigned)T), dim3(DD_), 0, stream, x, tri_r, tri_avg, (int)T);
        gemm(tri_avg, wnode + (size_t)r * DD_ * DD_, DD_, nullptr, nullptr, tri_ft, T, DD_, DD_, 1);
        gemm(tri_ft,  wtri  + (size_t)r * DD_ * DD_, DD_, nullptr, tri_msg, nullptr, T, DD_, DD_, 0);
        hipMemsetAsync(accum, 0, (size_t)N * DD_ * 4, stream);
        hipMemsetAsync(cnt,   0, (size_t)N * 4, stream);
        hipLaunchKernelGGL(scatter_k, dim3((unsigned)T), dim3(DD_), 0, stream, tri_msg, tri_r, accum, cnt, (int)T);
        hipLaunchKernelGGL(elu_k, dim3((unsigned)N), dim3(DD_), 0, stream, accum, cnt, u_f, u_b);
        gemm(u_b, wgate + DD_, 2 * DD_, nullptr, ug, nullptr, N, DD_, DD_, 0);
        hipLaunchKernelGGL(combine_k, dim3((unsigned)N), dim3(DD_), 0, stream,
                           xg, ug, gate_b, u_f, x_res,
                           src_f + (size_t)r * N * DD_, src_b + (size_t)r * N * DD_);
    }

    // ---- 3. transformer layer over the relation axis ----
    const long M = (long)RR_ * N;
    gemm(src_b, win, DD_, in_b, nullptr, qkv, M, 3 * DD_, DD_, 1);                 // qkv
    hipLaunchKernelGGL(attn_k, dim3((unsigned)((N * HH_ + 7) / 8)), dim3(256), 0, stream, qkv, o_b);
    gemm(o_b, wout, DD_, out_b, o_lin, nullptr, M, DD_, DD_, 0);                   // o @ out_W.T + out_b
    hipLaunchKernelGGL(ln1_k, dim3((unsigned)((M + 7) / 8)), dim3(256), 0, stream,
                       src_f, o_lin, ln1_g, ln1_b, x1_f, x1_b, M);
    gemm(x1_b, wlin1, DD_,     lin1_b, nullptr, h1_b, M, 2 * DD_, DD_,     2);     // gelu(lin1)
    gemm(h1_b, wlin2, 2 * DD_, lin2_b, ffp,     nullptr, M, DD_,  2 * DD_, 0);     // lin2
    hipLaunchKernelGGL(ln2_mean_k, dim3((unsigned)((N + 7) / 8)), dim3(256), 0, stream,
                       x1_f, ffp, ln2_g, ln2_b, out);
}
